// MetaLayer_9182640079111
// MI455X (gfx1250) — compile-verified, matrix operand order checked by reading, not run on hardware
//
#include <hip/hip_runtime.h>

// ---------------------------------------------------------------------------
// MetaLayer for MI455X (gfx1250): bf16 WMMA GEMM pipeline, f32 accumulate.
// Async global->LDS double-buffered GEMM, 256x128 workgroup tile,
// 64x64 per wave (4x2 waves), 16 v_wmma per K-step per wave.
// B=8, N=2048, D=1024, IN=128, EXP=4, BN=16384.
// ---------------------------------------------------------------------------

using bf16 = __bf16;
typedef __attribute__((ext_vector_type(16))) bf16  v16bf;
typedef __attribute__((ext_vector_type(8)))  float v8f;

#define BDIM   8
#define NDIM   2048
#define DDIM   1024
#define BN     (BDIM * NDIM)       // 16384
#define INDIM  128

__device__ __forceinline__ bf16 f2bf(float f) {
  unsigned int u = __builtin_bit_cast(unsigned int, f);
  u += 0x7FFFu + ((u >> 16) & 1u);                    // round-to-nearest-even
  unsigned short h = (unsigned short)(u >> 16);
  return __builtin_bit_cast(bf16, h);
}

__device__ __forceinline__ float gelu_tanh(float x) {
  float x3 = x * x * x;
  return 0.5f * x * (1.0f + tanhf(0.7978845608028654f * (x + 0.044715f * x3)));
}

struct Frag2 { uint4 lo, hi; };

__device__ __forceinline__ void async_b128(unsigned lds_off, const bf16* gptr) {
  asm volatile("global_load_async_to_lds_b128 %0, %1, off"
               :: "v"(lds_off), "v"(gptr) : "memory");
}
__device__ __forceinline__ void wait_async0() {
  asm volatile("s_wait_asynccnt 0x0" ::: "memory");
}

// ---------------------------------------------------------------------------
// GEMM: C[M,N'] = A[M,K] (row-major bf16) @ B[N',K]^T (row-major bf16)
// grid = (N'/128, M/256, batch), block = 256 (8 waves, 4x2, 64x64/wave).
// Fragment layout per CDNA5 ISA 7.12.2 (16-bit A 16x32: lane=(m,half),
// two contiguous 16B K-chunks); B operand symmetric over N'.
// ---------------------------------------------------------------------------
#define LDSS 56   // LDS row stride in bf16 (112B: 16B-aligned, conflict-free b128)

template <int ACT, bool BIAS, bool WF, bool WB>
__global__ __launch_bounds__(256) void gemm_bt(
    const bf16* __restrict__ A, const bf16* __restrict__ Bm,
    const float* __restrict__ bias,
    float* __restrict__ Cf, bf16* __restrict__ Cb,
    int K, int ldc,
    long long sA, long long sB, long long sC) {
  __shared__ bf16 As[2][256 * LDSS];   // 2 x 28672 B
  __shared__ bf16 Bs[2][128 * LDSS];   // 2 x 14336 B

  const int tid  = threadIdx.x;
  const int lane = tid & 31;
  const int w    = tid >> 5;
  const int wm   = w >> 1;          // 0..3  (64 rows each)
  const int wn   = w & 1;           // 0..1  (64 cols each)
  const int m    = lane & 15;       // row-in-16x16 for A/B frags; col for C
  const int h    = lane >> 4;       // half select

  const long long zb = blockIdx.z;
  const bf16* Ag = A  + zb * sA + (long long)blockIdx.y * 256 * K;
  const bf16* Bg = Bm + zb * sB + (long long)blockIdx.x * 128 * K;

  // Async-copy issue for one 32-wide K slab into buffer `buf`.
  auto issue = [&](int kk, int buf) {
#pragma unroll
    for (int i = 0; i < 4; ++i) {              // A: 256 rows x 4 chunks
      int c = tid + i * 256;
      int row = c >> 2, ch = (c & 3) * 8;
      async_b128((unsigned)(uintptr_t)&As[buf][row * LDSS + ch],
                 Ag + (long long)row * K + kk + ch);
    }
#pragma unroll
    for (int i = 0; i < 2; ++i) {              // B: 128 rows x 4 chunks
      int c = tid + i * 256;
      int row = c >> 2, ch = (c & 3) * 8;
      async_b128((unsigned)(uintptr_t)&Bs[buf][row * LDSS + ch],
                 Bg + (long long)row * K + kk + ch);
    }
  };

  v8f acc[4][4] = {};   // [ms][ns]

  issue(0, 0);
  int ib = 0;
  for (int kk = 0; kk < K; kk += 32, ib ^= 1) {
    wait_async0();        // own async loads for buf[ib] landed in LDS
    __syncthreads();      // everyone's landed; prev compute on buf[ib^1] done
    if (kk + 32 < K) issue(kk + 32, ib ^ 1);

    v16bf af[4];
#pragma unroll
    for (int ms = 0; ms < 4; ++ms) {
      int r = (wm * 64 + ms * 16 + m) * LDSS + h * 8;
      Frag2 t;
      t.lo = *reinterpret_cast<const uint4*>(&As[ib][r]);
      t.hi = *reinterpret_cast<const uint4*>(&As[ib][r + 16]);
      af[ms] = __builtin_bit_cast(v16bf, t);
    }
#pragma unroll
    for (int ns = 0; ns < 4; ++ns) {
      int r = (wn * 64 + ns * 16 + m) * LDSS + h * 8;
      Frag2 t;
      t.lo = *reinterpret_cast<const uint4*>(&Bs[ib][r]);
      t.hi = *reinterpret_cast<const uint4*>(&Bs[ib][r + 16]);
      v16bf bq = __builtin_bit_cast(v16bf, t);
#pragma unroll
      for (int ms = 0; ms < 4; ++ms)
        acc[ms][ns] = __builtin_amdgcn_wmma_f32_16x16x32_bf16(
            false, af[ms], false, bq, (short)0, acc[ms][ns], false, false);
    }
  }

  // Epilogue: lane holds C rows v+8h, col m within each 16x16 tile.
  const long long row0 = (long long)blockIdx.y * 256 + wm * 64 + 8 * h;
#pragma unroll
  for (int ns = 0; ns < 4; ++ns) {
    const int cg = blockIdx.x * 128 + wn * 64 + ns * 16 + m;
    const float bv = BIAS ? bias[cg] : 0.0f;
#pragma unroll
    for (int ms = 0; ms < 4; ++ms) {
      const long long base = zb * sC + (row0 + ms * 16) * (long long)ldc + cg;
      const v8f a = acc[ms][ns];
#pragma unroll
      for (int v = 0; v < 8; ++v) {
        float xv = a[v] + bv;
        if (ACT == 1) xv = gelu_tanh(xv);
        const long long idx = base + (long long)(v * ldc);
        if (WF) Cf[idx] = xv;
        if (WB) Cb[idx] = f2bf(xv);
      }
    }
  }
}

// ---------------------------------------------------------------------------
// LayerNorm over last dim (D=1024). One block (256 thr) per row.
// ---------------------------------------------------------------------------
__global__ __launch_bounds__(256) void ln_kernel(
    const float* __restrict__ in, const float* __restrict__ g,
    const float* __restrict__ b, float* __restrict__ outF,
    bf16* __restrict__ outB) {
  const int row = blockIdx.x;
  const int tid = threadIdx.x;
  const float* x = in + (long long)row * DDIM;
  float v[4], s = 0.f, ss = 0.f;
#pragma unroll
  for (int j = 0; j < 4; ++j) {
    v[j] = x[tid + j * 256];
    s += v[j];
    ss += v[j] * v[j];
  }
  __shared__ float r1[256], r2[256];
  r1[tid] = s; r2[tid] = ss;
  __syncthreads();
  for (int st = 128; st > 0; st >>= 1) {
    if (tid < st) { r1[tid] += r1[tid + st]; r2[tid] += r2[tid + st]; }
    __syncthreads();
  }
  float mean = r1[0] * (1.0f / DDIM);
  float var  = r2[0] * (1.0f / DDIM) - mean * mean;
  float rstd = rsqrtf(var + 1e-5f);
#pragma unroll
  for (int j = 0; j < 4; ++j) {
    int c = tid + j * 256;
    float y = (v[j] - mean) * rstd * g[c] + b[c];
    long long o = (long long)row * DDIM + c;
    if (outF) outF[o] = y;
    if (outB) outB[o] = f2bf(y);
  }
}

// ---------------------------------------------------------------------------
// Tiled transpose f32[R,C] -> bf16[C,R]; grid=(C/32, R/32, batch), block=(32,8)
// ---------------------------------------------------------------------------
__global__ void transpose_f2b(const float* __restrict__ src, bf16* __restrict__ dst,
                              int R, int C, long long sS, long long sD) {
  __shared__ float t[32][33];
  const long long zb = blockIdx.z;
  const int c0 = blockIdx.x * 32, r0 = blockIdx.y * 32;
#pragma unroll
  for (int j = 0; j < 4; ++j) {
    int r = r0 + threadIdx.y + j * 8;
    t[threadIdx.y + j * 8][threadIdx.x] =
        src[zb * sS + (long long)r * C + c0 + threadIdx.x];
  }
  __syncthreads();
#pragma unroll
  for (int j = 0; j < 4; ++j) {
    int cc = c0 + threadIdx.y + j * 8;
    int rr = r0 + threadIdx.x;
    dst[zb * sD + (long long)cc * R + rr] = f2bf(t[threadIdx.x][threadIdx.y + j * 8]);
  }
}

// ---------------------------------------------------------------------------
// Elementwise glue
// ---------------------------------------------------------------------------
__global__ void cvt_bf(const float* __restrict__ s, bf16* __restrict__ d, int n) {
  int i = blockIdx.x * 256 + threadIdx.x;
  if (i < n) d[i] = f2bf(s[i]);
}

// A = H[:, 0:128] as bf16  (n = BN*128)
__global__ void a_slice_k(const float* __restrict__ H, bf16* __restrict__ A, int n) {
  int i = blockIdx.x * 256 + threadIdx.x;
  if (i >= n) return;
  int r = i >> 7, c = i & 127;
  A[i] = f2bf(H[(long long)r * 256 + c]);
}

// S = silu(U)*V from G1=[U|V]  (n = BN*512)
__global__ void siluglu_k(const float* __restrict__ G1, bf16* __restrict__ S, int n) {
  int i = blockIdx.x * 256 + threadIdx.x;
  if (i >= n) return;
  int r = i >> 9, c = i & 511;
  float u  = G1[(long long)r * 1024 + c];
  float vv = G1[(long long)r * 1024 + 512 + c];
  float sg = 1.0f / (1.0f + expf(-u));
  S[i] = f2bf(u * sg * vv);
}

// R = (A2 - BB)*(2/IN); DH[:,128:256] = -R   (n = BN*128)
__global__ void resid_k(const float* __restrict__ A2, const float* __restrict__ H,
                        bf16* __restrict__ Rb, bf16* __restrict__ DH, int n) {
  int i = blockIdx.x * 256 + threadIdx.x;
  if (i >= n) return;
  int r = i >> 7, c = i & 127;
  float bb = H[(long long)r * 256 + 128 + c];
  float rv = (A2[i] - bb) * (2.0f / (float)INDIM);
  Rb[i] = f2bf(rv);
  DH[(long long)r * 256 + 128 + c] = f2bf(-rv);
}

// DG1 = [ DS*V*silu'(U) | DS*silu(U) ]   (n = BN*512)
__global__ void dg1_k(const float* __restrict__ DS, const float* __restrict__ G1,
                      bf16* __restrict__ DG1, int n) {
  int i = blockIdx.x * 256 + threadIdx.x;
  if (i >= n) return;
  int r = i >> 9, c = i & 511;
  float ds = DS[i];
  float u  = G1[(long long)r * 1024 + c];
  float vv = G1[(long long)r * 1024 + 512 + c];
  float sg = 1.0f / (1.0f + expf(-u));
  float si = u * sg;
  float dsi = sg * (1.0f + u * (1.0f - sg));
  DG1[(long long)r * 1024 + c]       = f2bf(ds * vv * dsi);
  DG1[(long long)r * 1024 + 512 + c] = f2bf(ds * si);
}

// Wd[b,q,z] = bf16(W_v[q,z] - grad_ff[b,q,z])   (n = B*D*D)
__global__ void wdelta_k(const float* __restrict__ Wv, const float* __restrict__ GF,
                         bf16* __restrict__ Wd, int n) {
  int i = blockIdx.x * 256 + threadIdx.x;
  if (i >= n) return;
  int qz = i & (DDIM * DDIM - 1);
  Wd[i] = f2bf(Wv[qz] - GF[i]);
}

// ---------------------------------------------------------------------------
// Launch
// ---------------------------------------------------------------------------
extern "C" void kernel_launch(void* const* d_in, const int* in_sizes, int n_in,
                              void* d_out, int out_size, void* d_ws, size_t ws_size,
                              hipStream_t stream) {
  (void)in_sizes; (void)n_in; (void)out_size; (void)ws_size;

  const float* x        = (const float*)d_in[0];
  const float* ln_in_g  = (const float*)d_in[1];
  const float* ln_in_b  = (const float*)d_in[2];
  const float* W_in     = (const float*)d_in[3];
  const float* W_v      = (const float*)d_in[4];
  const float* ll_W     = (const float*)d_in[5];
  const float* ll_b     = (const float*)d_in[6];
  const float* aff_in_W = (const float*)d_in[7];
  const float* aff_in_b = (const float*)d_in[8];
  const float* aff_out_W= (const float*)d_in[9];
  const float* aff_out_b= (const float*)d_in[10];
  const float* mlp1_W   = (const float*)d_in[11];
  const float* mlp1_b   = (const float*)d_in[12];
  const float* mlp2_W   = (const float*)d_in[13];
  const float* mlp2_b   = (const float*)d_in[14];
  const float* ln_out_g = (const float*)d_in[15];
  const float* ln_out_b = (const float*)d_in[16];
  float* out = (float*)d_out;

  // ---- workspace bump allocator (256B aligned) ----
  size_t off = 0;
  auto alloc = [&](size_t bytes) -> void* {
    void* p = (char*)d_ws + off;
    off += (bytes + 255) & ~(size_t)255;
    return p;
  };
  const size_t f4 = sizeof(float), b2 = sizeof(bf16);

  float* xn_f  = (float*)alloc((size_t)BN * DDIM * f4);       // reused as Hout
  bf16*  xn_b  = (bf16*) alloc((size_t)BN * DDIM * b2);
  bf16*  xnT_b = (bf16*) alloc((size_t)BDIM * DDIM * NDIM * b2);
  bf16*  xff_b = (bf16*) alloc((size_t)BN * DDIM * b2);
  bf16*  Win_b = (bf16*) alloc((size_t)DDIM * DDIM * b2);
  bf16*  llW_b = (bf16*) alloc((size_t)256 * DDIM * b2);
  bf16*  llWt_b= (bf16*) alloc((size_t)DDIM * 256 * b2);
  bf16*  ai_b  = (bf16*) alloc((size_t)1024 * 128 * b2);
  bf16*  aiT_b = (bf16*) alloc((size_t)128 * 1024 * b2);
  bf16*  ao_b  = (bf16*) alloc((size_t)128 * 512 * b2);
  bf16*  aoT_b = (bf16*) alloc((size_t)512 * 128 * b2);
  bf16*  m1_b  = (bf16*) alloc((size_t)2048 * 1024 * b2);
  bf16*  m2_b  = (bf16*) alloc((size_t)1024 * 2048 * b2);
  float* H_f   = (float*)alloc((size_t)BN * 256 * f4);
  bf16*  A_b   = (bf16*) alloc((size_t)BN * 128 * b2);
  float* G1_f  = (float*)alloc((size_t)BN * 1024 * f4);       // reused as grad_ff
  bf16*  S_b   = (bf16*) alloc((size_t)BN * 512 * b2);
  float* A2_f  = (float*)alloc((size_t)BN * 128 * f4);
  bf16*  R_b   = (bf16*) alloc((size_t)BN * 128 * b2);
  float* DS_f  = (float*)alloc((size_t)BN * 512 * f4);
  bf16*  DG1_b = (bf16*) alloc((size_t)BN * 1024 * b2);
  bf16*  DH_b  = (bf16*) alloc((size_t)BN * 256 * b2);
  float* G_f   = (float*)alloc((size_t)BN * DDIM * f4);
  bf16*  gT_b  = (bf16*) alloc((size_t)BDIM * DDIM * NDIM * b2);
  bf16*  Wd_b  = (bf16*) alloc((size_t)BDIM * DDIM * DDIM * b2);
  bf16*  xu_b  = (bf16*) alloc((size_t)BN * DDIM * b2);
  bf16*  H1_b  = (bf16*) alloc((size_t)BN * 2048 * b2);
  float* GF_f   = G1_f;    // alias: G1 dead before grad_ff is produced
  float* Hout_f = xn_f;    // alias: xn_f dead after xnT transpose

  const dim3 blk256(256), blkT(32, 8);
  auto g1d = [](int n) { return dim3((n + 255) / 256); };

  // 1) input layernorm -> xn_f (for transpose) + xn_b (GEMM operand)
  ln_kernel<<<dim3(BN), blk256, 0, stream>>>(x, ln_in_g, ln_in_b, xn_f, xn_b);

  // 2) weight conversions (bf16) and transposes
  cvt_bf<<<g1d(DDIM * DDIM), blk256, 0, stream>>>(W_in, Win_b, DDIM * DDIM);
  cvt_bf<<<g1d(256 * DDIM), blk256, 0, stream>>>(ll_W, llW_b, 256 * DDIM);
  cvt_bf<<<g1d(1024 * 128), blk256, 0, stream>>>(aff_in_W, ai_b, 1024 * 128);
  cvt_bf<<<g1d(128 * 512), blk256, 0, stream>>>(aff_out_W, ao_b, 128 * 512);
  cvt_bf<<<g1d(2048 * 1024), blk256, 0, stream>>>(mlp1_W, m1_b, 2048 * 1024);
  cvt_bf<<<g1d(1024 * 2048), blk256, 0, stream>>>(mlp2_W, m2_b, 1024 * 2048);
  transpose_f2b<<<dim3(32, 8, 1), blkT, 0, stream>>>(ll_W, llWt_b, 256, 1024, 0, 0);
  transpose_f2b<<<dim3(4, 32, 1), blkT, 0, stream>>>(aff_in_W, aiT_b, 1024, 128, 0, 0);
  transpose_f2b<<<dim3(16, 4, 1), blkT, 0, stream>>>(aff_out_W, aoT_b, 128, 512, 0, 0);

  // 3) xnT (per-sample transpose for fast-weight einsum)
  transpose_f2b<<<dim3(DDIM / 32, NDIM / 32, BDIM), blkT, 0, stream>>>(
      xn_f, xnT_b, NDIM, DDIM, (long long)NDIM * DDIM, (long long)DDIM * NDIM);

  // 4) x_ff = xn @ W_in^T                                   [BN,1024] bf16
  gemm_bt<0, false, false, true><<<dim3(8, BN / 256, 1), blk256, 0, stream>>>(
      xn_b, Win_b, nullptr, nullptr, xff_b, DDIM, DDIM, 0, 0, 0);

  // 5) H = x_ff @ ll_W^T + ll_b                             [BN,256] f32
  gemm_bt<0, true, true, false><<<dim3(2, BN / 256, 1), blk256, 0, stream>>>(
      xff_b, llW_b, ll_b, H_f, nullptr, DDIM, 256, 0, 0, 0);
  a_slice_k<<<g1d(BN * 128), blk256, 0, stream>>>(H_f, A_b, BN * 128);

  // 6) G1 = A @ aff_in_W^T + aff_in_b                       [BN,1024] f32
  gemm_bt<0, true, true, false><<<dim3(8, BN / 256, 1), blk256, 0, stream>>>(
      A_b, ai_b, aff_in_b, G1_f, nullptr, 128, 1024, 0, 0, 0);
  siluglu_k<<<g1d(BN * 512), blk256, 0, stream>>>(G1_f, S_b, BN * 512);

  // 7) A2 = S @ aff_out_W^T + aff_out_b                     [BN,128] f32
  gemm_bt<0, true, true, false><<<dim3(1, BN / 256, 1), blk256, 0, stream>>>(
      S_b, ao_b, aff_out_b, A2_f, nullptr, 512, 128, 0, 0, 0);
  resid_k<<<g1d(BN * 128), blk256, 0, stream>>>(A2_f, H_f, R_b, DH_b, BN * 128);

  // 8) DS = R @ aff_out_W  (= R @ aoT^T)                    [BN,512] f32
  gemm_bt<0, false, true, false><<<dim3(4, BN / 256, 1), blk256, 0, stream>>>(
      R_b, aoT_b, nullptr, DS_f, nullptr, 128, 512, 0, 0, 0);
  dg1_k<<<g1d(BN * 512), blk256, 0, stream>>>(DS_f, G1_f, DG1_b, BN * 512);

  // 9) DA = DG1 @ aff_in_W  (= DG1 @ aiT^T) -> DH[:,0:128]  (bf16, ldc=256)
  gemm_bt<0, false, false, true><<<dim3(1, BN / 256, 1), blk256, 0, stream>>>(
      DG1_b, aiT_b, nullptr, nullptr, DH_b, 1024, 256, 0, 0, 0);

  // 10) g = DH @ ll_W  (= DH @ llWt^T)                      [BN,1024] f32
  gemm_bt<0, false, true, false><<<dim3(8, BN / 256, 1), blk256, 0, stream>>>(
      DH_b, llWt_b, nullptr, G_f, nullptr, 256, DDIM, 0, 0, 0);
  transpose_f2b<<<dim3(DDIM / 32, NDIM / 32, BDIM), blkT, 0, stream>>>(
      G_f, gT_b, NDIM, DDIM, (long long)NDIM * DDIM, (long long)DDIM * NDIM);

  // 11) grad_ff[b] = g[b]^T @ xn[b]  (= gT @ xnT^T)         [B,1024,1024] f32
  gemm_bt<0, false, true, false><<<dim3(8, 4, BDIM), blk256, 0, stream>>>(
      gT_b, xnT_b, nullptr, GF_f, nullptr, NDIM, DDIM,
      (long long)DDIM * NDIM, (long long)DDIM * NDIM, (long long)DDIM * DDIM);
  wdelta_k<<<g1d(BDIM * DDIM * DDIM), blk256, 0, stream>>>(W_v, GF_f, Wd_b,
                                                           BDIM * DDIM * DDIM);

  // 12) x_u[b] = xn[b] @ (W_v - grad_ff[b])^T               [B,2048,1024] bf16
  gemm_bt<0, false, false, true><<<dim3(8, 8, BDIM), blk256, 0, stream>>>(
      xn_b, Wd_b, nullptr, nullptr, xu_b, DDIM, DDIM,
      (long long)NDIM * DDIM, (long long)DDIM * DDIM, (long long)NDIM * DDIM);

  // 13) H1 = gelu(x_u @ mlp1_W^T + b1)                      [BN,2048] bf16
  gemm_bt<1, true, false, true><<<dim3(16, BN / 256, 1), blk256, 0, stream>>>(
      xu_b, m1_b, mlp1_b, nullptr, H1_b, DDIM, 2048, 0, 0, 0);

  // 14) Hout = H1 @ mlp2_W^T + b2                           [BN,1024] f32
  gemm_bt<0, true, true, false><<<dim3(8, BN / 256, 1), blk256, 0, stream>>>(
      H1_b, m2_b, mlp2_b, Hout_f, nullptr, 2048, DDIM, 0, 0, 0);

  // 15) output layernorm -> d_out (f32)
  ln_kernel<<<dim3(BN), blk256, 0, stream>>>(Hout_f, ln_out_g, ln_out_b, out, nullptr);
}
